// SingleLayerGLIFRSNN_51470888075448
// MI455X (gfx1250) — matrix operand order repaired and checked by reading, not compiled, not run on hardware
//
#include <hip/hip_runtime.h>
#include <hip/hip_bf16.h>

// ---------------- model constants ----------------
constexpr int   BATCH = 32;
constexpr int   NN    = 1000;     // neurons
constexpr int   NPAD  = 1008;     // padded neuron dim (63 * 16)
constexpr int   NT    = 63;       // n-tiles of 16
constexpr int   INDIM = 128;
constexpr int   KPAD  = 1152;     // 1000 (s) + 24 pad + 128 (x)  = 36 * 32
constexpr int   KT    = 36;       // k-tiles of 32
constexpr int   KT_REC= 32;       // k-tiles belonging to W_rec (k < 1024)
constexpr int   XOFF  = 1024;     // column where x starts in A / W_in starts in B
constexpr int   TSTEPS= 1000;
constexpr int   OUTD  = 20;
constexpr int   TSTART= 200;      // int(1000 * (1 - 0.8))
constexpr float V_TH    = -45.0f;
constexpr float V_RESET = -60.0f;
constexpr float A_V     = 0.95122942450071400f;  // exp(-1/20)
constexpr float OM_A_V  = 0.04877057549928600f;
constexpr float A_SYN   = 0.81873075307798182f;  // exp(-1/5)
constexpr float A_READ  = 0.95122942450071400f;  // exp(-1/20)
constexpr float OM_A_RD = 0.04877057549928600f;

typedef __attribute__((ext_vector_type(16))) _Float16 v16h;
typedef __attribute__((ext_vector_type(8)))  _Float16 v8h;
typedef __attribute__((ext_vector_type(8)))  float    v8f;

// ---------------- workspace layout (bytes) ----------------
constexpr size_t BP_BYTES  = (size_t)KT * NT * 32 * 16 * 2;   // packed B frags: 2,322,432
constexpr size_t A_BYTES   = (size_t)BATCH * KPAD * 2;        // 73,728
constexpr size_t ST_BYTES  = (size_t)BATCH * NPAD * 4;        // 129,024
constexpr size_t OFF_BP    = 0;
constexpr size_t OFF_A0    = OFF_BP + BP_BYTES;
constexpr size_t OFF_A1    = OFF_A0 + A_BYTES;
constexpr size_t OFF_V     = OFF_A1 + A_BYTES;
constexpr size_t OFF_IASC  = OFF_V    + ST_BYTES;
constexpr size_t OFF_PSC   = OFF_IASC + ST_BYTES;
constexpr size_t OFF_F     = OFF_PSC  + ST_BYTES;
constexpr size_t OFF_ACC   = OFF_F    + ST_BYTES;

// =====================================================================
// Pack B = [W_rec ; 0 ; W_in^T] (KPAD x NPAD, f16) into per-lane WMMA
// B-fragment layout:
//   Bp[((kt*NT+nt)*32 + lane)*16 + s] = B[kt*32 + (s>>3)*16 + (lane>>4)*8 + (s&7)][nt*16 + (lane&15)]
// =====================================================================
__global__ void rsnn_pack_b(const float* __restrict__ W_rec,   // (1000, 1000) (pre, post)
                            const float* __restrict__ W_in,    // (1000, 128)
                            _Float16* __restrict__ Bp)
{
    const int total = KT * NT * 32 * 16;
    for (int h = blockIdx.x * blockDim.x + threadIdx.x; h < total;
         h += gridDim.x * blockDim.x) {
        int s    = h & 15;
        int lane = (h >> 4) & 31;
        int tile = h >> 9;
        int nt   = tile % NT;
        int kt   = tile / NT;
        int k    = kt * 32 + ((s >> 3) << 4) + ((lane >> 4) << 3) + (s & 7);
        int n    = nt * 16 + (lane & 15);
        float v  = 0.0f;
        if (n < NN) {
            if (k < NN)                  v = W_rec[(size_t)k * NN + n];
            else if (k >= XOFF && k < XOFF + INDIM)
                                         v = W_in[(size_t)n * INDIM + (k - XOFF)];
        }
        Bp[h] = (_Float16)v;
    }
}

// =====================================================================
// Init: states + A double buffer (A0 carries x[0]*in_scale; spikes = 0)
// =====================================================================
__global__ void rsnn_init(const float* __restrict__ x,          // (T, B, 128)
                          const float* __restrict__ in_scale_p,
                          float* __restrict__ v, float* __restrict__ iasc,
                          float* __restrict__ psc, float* __restrict__ f,
                          float* __restrict__ acc,
                          _Float16* __restrict__ A0, _Float16* __restrict__ A1)
{
    const float in_scale = *in_scale_p;
    const int gsz = gridDim.x * blockDim.x;
    const int gid = blockIdx.x * blockDim.x + threadIdx.x;
    for (int i = gid; i < BATCH * NPAD; i += gsz) {
        v[i]    = V_RESET;
        iasc[i] = 0.0f;
        psc[i]  = 0.0f;
        f[i]    = 0.0f;
        acc[i]  = 0.0f;
    }
    for (int i = gid; i < BATCH * KPAD; i += gsz) {
        int b = i / KPAD, c = i % KPAD;
        float val = 0.0f;
        if (c >= XOFF && c < XOFF + INDIM)
            val = x[(size_t)b * INDIM + (c - XOFF)] * in_scale;
        A0[i] = (_Float16)val;
        A1[i] = (_Float16)0.0f;   // pad columns of A1 must stay zero forever
    }
}

// =====================================================================
// One timestep. grid = NT blocks x 64 threads (wave w -> batch tile w).
// B fragments for this block's n-tile are staged once into LDS and
// shared by both waves; GEMM is two straight-line WMMA loops
// (recurrent accumulator, input accumulator).
// =====================================================================
__global__ void __launch_bounds__(64)
rsnn_step(const float* __restrict__ x,
          const float* __restrict__ asc_amps,
          const float* __restrict__ k_decay,
          const float* __restrict__ in_scale_p,
          const _Float16* __restrict__ Bp,
          const _Float16* __restrict__ A_cur,
          _Float16* __restrict__ A_next,
          float* __restrict__ v_arr, float* __restrict__ iasc_arr,
          float* __restrict__ psc_arr, float* __restrict__ f_arr,
          float* __restrict__ acc_arr,
          int t)
{
    __shared__ _Float16 Bsh[KT * 32 * 16];     // 36 KB

    const int tid  = threadIdx.x;
    const int lane = tid & 31;
    const int wave = tid >> 5;                 // 0..1 -> batch tile
    const int nt   = blockIdx.x;               // 0..62
    const int m0   = wave * 16;

    // ---- early prefetch of the x[t+1] block this block will consume ----
    if (wave == 0 && nt < (INDIM / 16) && t + 1 < TSTEPS)
        __builtin_prefetch(
            (const void*)&x[((size_t)(t + 1) * BATCH + lane) * INDIM + nt * 16], 0, 1);

    // ---- cooperative stage of this n-tile's B fragments into LDS ----
    // per k-tile: 32 lanes * 32 B = 1 KB contiguous in Bp
    {
        const _Float16* src = Bp + (size_t)nt * 512;     // tile (0, nt)
        #pragma unroll 4
        for (int c = tid; c < KT * 64; c += 64) {        // 16-B chunks
            int kt  = c >> 6;
            int off = (c & 63) * 8;
            *(v8h*)(Bsh + kt * 512 + off) =
                *(const v8h*)(src + (size_t)kt * NT * 512 + off);
        }
    }
    __syncthreads();

    // ---- per-lane A row pointer (row = batch), 16-bit WMMA A layout ----
    const int arow  = m0 + (lane & 15);
    const int kbOff = (lane >> 4) * 8;
    const _Float16* Arow  = A_cur + (size_t)arow * KPAD;
    const _Float16* BshLn = Bsh + lane * 16;

    v8f c_rec = {};   // W_rec contributions (k-tiles 0..31)
    v8f c_in  = {};   // W_in  contributions (k-tiles 32..35)

    #pragma unroll 8
    for (int kt = 0; kt < KT_REC; ++kt) {
        v8h lo = *(const v8h*)(Arow + kt * 32 + kbOff);
        v8h hi = *(const v8h*)(Arow + kt * 32 + kbOff + 16);
        v16h afrag;
        #pragma unroll
        for (int i = 0; i < 8; ++i) { afrag[i] = lo[i]; afrag[i + 8] = hi[i]; }
        v16h bfrag = *(const v16h*)(BshLn + kt * 512);
        c_rec = __builtin_amdgcn_wmma_f32_16x16x32_f16(
            false, afrag, false, bfrag, (short)0, c_rec, false, false);
    }
    #pragma unroll
    for (int kt = KT_REC; kt < KT; ++kt) {
        v8h lo = *(const v8h*)(Arow + kt * 32 + kbOff);
        v8h hi = *(const v8h*)(Arow + kt * 32 + kbOff + 16);
        v16h afrag;
        #pragma unroll
        for (int i = 0; i < 8; ++i) { afrag[i] = lo[i]; afrag[i + 8] = hi[i]; }
        v16h bfrag = *(const v16h*)(BshLn + kt * 512);
        c_in = __builtin_amdgcn_wmma_f32_16x16x32_f16(
            false, afrag, false, bfrag, (short)0, c_in, false, false);
    }

    // ---- elementwise neuron update (D layout: lane -> n, VGPR r -> batch) ----
    const int n   = nt * 16 + (lane & 15);
    const int nc  = n < NN ? n : NN - 1;
    const float a_asc = __expf(-k_decay[nc]);
    const float amp   = asc_amps[nc];
    const int   bhi   = (lane >= 16) ? 8 : 0;

    #pragma unroll
    for (int r = 0; r < 8; ++r) {
        const int b   = m0 + bhi + r;
        const size_t idx = (size_t)b * NPAD + n;

        float psc = A_SYN * psc_arr[idx] + c_rec[r];
        float Itot = c_in[r] + psc + iasc_arr[idx];
        float v = V_RESET + A_V * (v_arr[idx] - V_RESET) + OM_A_V * Itot;
        float s = (v - V_TH >= 0.0f) ? 1.0f : 0.0f;
        v = v - (v - V_RESET) * s;
        float ia = a_asc * iasc_arr[idx] + amp * s;

        psc_arr[idx]  = psc;
        v_arr[idx]    = v;
        iasc_arr[idx] = ia;

        float fv = (t == 0) ? s : (A_READ * f_arr[idx] + OM_A_RD * s);
        f_arr[idx] = fv;
        if (t >= TSTART) acc_arr[idx] += fv;

        if (n < NN) A_next[(size_t)b * KPAD + n] = (_Float16)s;
    }

    // ---- stage x[t+1] * in_scale into A_next input columns ----
    if (wave == 0 && nt < (INDIM / 16)) {
        const float in_scale = *in_scale_p;
        for (int i = lane; i < 16 * BATCH; i += 32) {
            int bb = i >> 4;
            int dd = nt * 16 + (i & 15);
            float xv = 0.0f;
            if (t + 1 < TSTEPS)
                xv = x[((size_t)(t + 1) * BATCH + bb) * INDIM + dd] * in_scale;
            A_next[(size_t)bb * KPAD + XOFF + dd] = (_Float16)xv;
        }
    }
}

// =====================================================================
// Readout: out[b][o] = b_out[o] + out_scale * (acc[b]/800) . W_out[o]
// =====================================================================
__global__ void rsnn_readout(const float* __restrict__ acc,
                             const float* __restrict__ W_out,   // (20, 1000)
                             const float* __restrict__ b_out,   // (20,)
                             const float* __restrict__ out_scale_p,
                             float* __restrict__ out)           // (32, 20)
{
    int tid = blockIdx.x * blockDim.x + threadIdx.x;
    if (tid >= BATCH * OUTD) return;
    int b = tid / OUTD, o = tid % OUTD;
    const float scale = (*out_scale_p) / (float)(TSTEPS - TSTART);
    float sum = 0.0f;
    const float* a = acc + (size_t)b * NPAD;
    const float* w = W_out + (size_t)o * NN;
    for (int n = 0; n < NN; ++n) sum += a[n] * w[n];
    out[(size_t)b * OUTD + o] = b_out[o] + scale * sum;
}

// =====================================================================
extern "C" void kernel_launch(void* const* d_in, const int* in_sizes, int n_in,
                              void* d_out, int out_size, void* d_ws, size_t ws_size,
                              hipStream_t stream)
{
    const float* x         = (const float*)d_in[0];
    const float* W_in      = (const float*)d_in[1];
    const float* W_rec     = (const float*)d_in[2];
    const float* asc_amps  = (const float*)d_in[3];
    const float* k_decay   = (const float*)d_in[4];
    const float* W_out     = (const float*)d_in[5];
    const float* b_out     = (const float*)d_in[6];
    const float* in_scale  = (const float*)d_in[7];
    const float* out_scale = (const float*)d_in[8];

    char* ws = (char*)d_ws;
    _Float16* Bp   = (_Float16*)(ws + OFF_BP);
    _Float16* A0   = (_Float16*)(ws + OFF_A0);
    _Float16* A1   = (_Float16*)(ws + OFF_A1);
    float*    v    = (float*)(ws + OFF_V);
    float*    iasc = (float*)(ws + OFF_IASC);
    float*    psc  = (float*)(ws + OFF_PSC);
    float*    f    = (float*)(ws + OFF_F);
    float*    acc  = (float*)(ws + OFF_ACC);

    // 1) pack B fragments (f16, WMMA lane layout)
    {
        int total = KT * NT * 32 * 16;
        int blocks = (total + 255) / 256;
        rsnn_pack_b<<<blocks, 256, 0, stream>>>(W_rec, W_in, Bp);
    }
    // 2) init state + A double buffer
    rsnn_init<<<256, 256, 0, stream>>>(x, in_scale, v, iasc, psc, f, acc, A0, A1);

    // 3) 1000 stream-ordered timesteps (graph-captured)
    for (int t = 0; t < TSTEPS; ++t) {
        const _Float16* Ac = (t & 1) ? A1 : A0;
        _Float16*       An = (t & 1) ? A0 : A1;
        rsnn_step<<<NT, 64, 0, stream>>>(x, asc_amps, k_decay, in_scale,
                                         Bp, Ac, An, v, iasc, psc, f, acc, t);
    }

    // 4) readout
    {
        int total = BATCH * OUTD;
        int blocks = (total + 63) / 64;
        rsnn_readout<<<blocks, 64, 0, stream>>>(acc, W_out, b_out, out_scale,
                                                (float*)d_out);
    }
}
